// EnhancedGNNBranch_4621384810730
// MI455X (gfx1250) — compile-verified, matrix operand order checked by reading, not run on hardware
//
#include <hip/hip_runtime.h>
#include <hip/hip_bf16.h>

// ---------------------------------------------------------------------------
// CDNA5 (gfx1250) implementation of the EnhancedGNNBranch reference.
// Dense projections run on v_wmma_f32_16x16x32_f16 with f16 operands that are
// pre-converted/pre-transposed into WMMA-friendly layouts so the GEMM inner
// loop is pure global_load_b128 + v_wmma (no predication, EXEC all-ones).
// Edge scatter-softmax uses ordered-uint atomic max + float atomic adds.
// TopK pooling is a per-graph bitonic sort in LDS (512-thread workgroups).
// ---------------------------------------------------------------------------

typedef __attribute__((ext_vector_type(16))) _Float16 v16h;
typedef __attribute__((ext_vector_type(8)))  _Float16 v8h;
typedef __attribute__((ext_vector_type(8)))  float    v8f;

#define CDIV(a, b) (((a) + (b) - 1) / (b))

namespace {
constexpr int BGRAPH = 32;               // graphs
constexpr int NPG0   = 512;              // nodes per graph (level 0)
constexpr int NNODES = BGRAPH * NPG0;    // 16384
constexpr int NEDGE  = BGRAPH * 8192;    // 262144
constexpr int DINF   = 200;
constexpr int HEADS  = 4;
constexpr int KP1    = 410;              // ceil(0.8*512)
constexpr int KP2    = 246;              // ceil(0.6*410)
constexpr int KPAD0  = 224;              // DIN padded to multiple of 32
}

// ---------------------------------------------------------------------------
// Operand conversion kernels (fp32 -> f16, K padded to multiple of 32).
// ---------------------------------------------------------------------------
__global__ void cvt_a_f16(const float* __restrict__ A, _Float16* __restrict__ Ah,
                          int M, int K, int Kp) {
  int i = blockIdx.x * blockDim.x + threadIdx.x;
  if (i >= M * Kp) return;
  int r = i / Kp, c = i - r * Kp;
  Ah[i] = (c < K) ? (_Float16)A[(size_t)r * K + c] : (_Float16)0.f;
}
// W[K,N] fp32 -> Wt[N,Kp] f16 (transposed so B columns are K-contiguous)
__global__ void cvt_w_f16_t(const float* __restrict__ W, _Float16* __restrict__ Wt,
                            int K, int N, int Kp) {
  int i = blockIdx.x * blockDim.x + threadIdx.x;
  if (i >= N * Kp) return;
  int n = i / Kp, k = i - n * Kp;
  Wt[i] = (k < K) ? (_Float16)W[(size_t)k * N + n] : (_Float16)0.f;
}

// ---------------------------------------------------------------------------
// WMMA GEMM: C[M,N] = A[M,K] @ W[K,N] (+ bias[N] if non-null).
// Operands: Ah f16 [M,Kp] row-major; Bt f16 [N,Kp] row-major (W transposed).
// One wave computes a 16(M) x 32(N) tile: A loaded once per k-step feeds two
// v_wmma_f32_16x16x32_f16. Per the CDNA5 16-bit operand maps, lane&15 selects
// the M row (A) / N col (B); lane>>4 selects the K-half interleave
// {0..7,16..23} vs {8..15,24..31} -> each lane's 16 elements are two
// contiguous 16B runs => global_load_b128 only, no predication.
// Requires: M % 16 == 0, N % 32 == 0, Kp % 32 == 0 (true for every call).
// ---------------------------------------------------------------------------
__global__ void wmma_gemm_f16(const _Float16* __restrict__ Ah,
                              const _Float16* __restrict__ Bt,
                              const float* __restrict__ bias,
                              float* __restrict__ C,
                              int M, int Nc, int Kp) {
  const int lane  = threadIdx.x & 31;
  const int khalf = lane >> 4;          // 0 or 1
  const int l16   = lane & 15;
  const int mrow  = blockIdx.y * 16 + l16;
  const int n0    = blockIdx.x * 32;

  const _Float16* ap  = Ah + (size_t)mrow * Kp + khalf * 8;
  const _Float16* bp0 = Bt + (size_t)(n0 + l16) * Kp + khalf * 8;
  const _Float16* bp1 = Bt + (size_t)(n0 + 16 + l16) * Kp + khalf * 8;

  v8f acc0 = {}, acc1 = {};
  for (int k0 = 0; k0 < Kp; k0 += 32) {
    __builtin_prefetch(ap + 32, 0, 1);     // global_prefetch_b8 (next k-step)
    __builtin_prefetch(bp0 + 32, 0, 1);
    __builtin_prefetch(bp1 + 32, 0, 1);
    v8h alo = *(const v8h*)(ap);           // K = k0 + khalf*8 + {0..7}
    v8h ahi = *(const v8h*)(ap + 16);      // K = k0 + 16 + khalf*8 + {0..7}
    v8h b0l = *(const v8h*)(bp0);
    v8h b0h = *(const v8h*)(bp0 + 16);
    v8h b1l = *(const v8h*)(bp1);
    v8h b1h = *(const v8h*)(bp1 + 16);
    v16h a  = __builtin_shufflevector(alo, ahi, 0,1,2,3,4,5,6,7,8,9,10,11,12,13,14,15);
    v16h b0 = __builtin_shufflevector(b0l, b0h, 0,1,2,3,4,5,6,7,8,9,10,11,12,13,14,15);
    v16h b1 = __builtin_shufflevector(b1l, b1h, 0,1,2,3,4,5,6,7,8,9,10,11,12,13,14,15);
    acc0 = __builtin_amdgcn_wmma_f32_16x16x32_f16(false, a, false, b0, (short)0, acc0, false, false);
    acc1 = __builtin_amdgcn_wmma_f32_16x16x32_f16(false, a, false, b1, (short)0, acc1, false, false);
    ap += 32; bp0 += 32; bp1 += 32;
  }
  // C/D layout: VGPR v holds row (tile_m*16 + khalf*8 + v), col = lane&15.
  const int rb = blockIdx.y * 16 + khalf * 8;
  const int c0 = n0 + l16, c1 = n0 + 16 + l16;
  const float bv0 = bias ? bias[c0] : 0.f;
  const float bv1 = bias ? bias[c1] : 0.f;
#pragma unroll
  for (int v = 0; v < 8; ++v) {
    C[(size_t)(rb + v) * Nc + c0] = acc0[v] + bv0;
    C[(size_t)(rb + v) * Nc + c1] = acc1[v] + bv1;
  }
}

// ---------------------------------------------------------------------------
// Utility fills
// ---------------------------------------------------------------------------
__global__ void fill_f32(float* p, float v, int n) {
  int i = blockIdx.x * blockDim.x + threadIdx.x;
  if (i < n) p[i] = v;
}
__global__ void fill_u32(unsigned* p, unsigned v, int n) {
  int i = blockIdx.x * blockDim.x + threadIdx.x;
  if (i < n) p[i] = v;
}
__global__ void fill_i32(int* p, int v, int n) {
  int i = blockIdx.x * blockDim.x + threadIdx.x;
  if (i < n) p[i] = v;
}

// Monotone float <-> uint key for atomic max on floats.
__device__ __forceinline__ unsigned f32key(float f) {
  unsigned u = __float_as_uint(f);
  return (u & 0x80000000u) ? ~u : (u | 0x80000000u);
}
__device__ __forceinline__ float keyf32(unsigned k) {
  return (k & 0x80000000u) ? __uint_as_float(k & 0x7FFFFFFFu)
                           : __uint_as_float(~k);
}

// asrc[n,h] = sum_c z[n,h,c]*a_s[h,c]; same for adst.
__global__ void attn_coef(const float* __restrict__ z,
                          const float* __restrict__ a_s,
                          const float* __restrict__ a_d,
                          float* __restrict__ asrc, float* __restrict__ adst,
                          int n, int H, int C) {
  int gid = blockIdx.x * blockDim.x + threadIdx.x;
  if (gid >= n * H) return;
  int node = gid / H, h = gid - node * H;
  const float* zp = z + (size_t)node * H * C + (size_t)h * C;
  float s1 = 0.f, s2 = 0.f;
  for (int c = 0; c < C; ++c) {
    float zv = zp[c];
    s1 += zv * a_s[h * C + c];
    s2 += zv * a_d[h * C + c];
  }
  asrc[gid] = s1;
  adst[gid] = s2;
}

__device__ __forceinline__ float leaky02(float e) { return e > 0.f ? e : 0.2f * e; }

// Pass 1: segment max of leaky_relu(asrc[s]+adst[d]) over destinations.
__global__ void gat_edge_max(const int* __restrict__ src, const int* __restrict__ dst,
                             const int* __restrict__ em,
                             const float* __restrict__ asrc, const float* __restrict__ adst,
                             unsigned* __restrict__ emaxk, int E, int n, int H) {
  int gid = blockIdx.x * blockDim.x + threadIdx.x;
  int T = E + n;
  if (gid >= T * H) return;
  int id = gid / H, h = gid - id * H;
  int s, d, m;
  if (id < E) { s = src[id]; d = dst[id]; m = em ? em[id] : 1; }
  else        { s = d = id - E; m = 1; }          // self loop
  if (!m) return;
  float e = leaky02(asrc[s * H + h] + adst[d * H + h]);
  atomicMax(&emaxk[d * H + h], f32key(e));
}

// Pass 2: den[d,h] += exp(e - emax[d,h])
__global__ void gat_edge_expsum(const int* __restrict__ src, const int* __restrict__ dst,
                                const int* __restrict__ em,
                                const float* __restrict__ asrc, const float* __restrict__ adst,
                                const unsigned* __restrict__ emaxk,
                                float* __restrict__ den, int E, int n, int H) {
  int gid = blockIdx.x * blockDim.x + threadIdx.x;
  int T = E + n;
  if (gid >= T * H) return;
  int id = gid / H, h = gid - id * H;
  int s, d, m;
  if (id < E) { s = src[id]; d = dst[id]; m = em ? em[id] : 1; }
  else        { s = d = id - E; m = 1; }
  if (!m) return;
  float e = leaky02(asrc[s * H + h] + adst[d * H + h]);
  float ex = expf(e - keyf32(emaxk[d * H + h]));
  atomicAdd(&den[d * H + h], ex);
}

// Pass 3: alpha = ex/den -> attn output; hm[d,c] += (alpha/H)*z[s,h,c].
// (Head-mean folded into aggregation; GAT bias cancelled by following BN.)
__global__ void gat_edge_aggr(const int* __restrict__ src, const int* __restrict__ dst,
                              const int* __restrict__ em,
                              const float* __restrict__ asrc, const float* __restrict__ adst,
                              const unsigned* __restrict__ emaxk,
                              const float* __restrict__ den,
                              const float* __restrict__ z,
                              float* __restrict__ hm, float* __restrict__ attn,
                              int E, int n, int H, int C) {
  int gid = blockIdx.x * blockDim.x + threadIdx.x;
  int T = E + n;
  if (gid >= T * H) return;
  int id = gid / H, h = gid - id * H;
  int s, d, m;
  if (id < E) { s = src[id]; d = dst[id]; m = em ? em[id] : 1; }
  else        { s = d = id - E; m = 1; }
  if (!m) { attn[gid] = 0.f; return; }
  float e  = leaky02(asrc[s * H + h] + adst[d * H + h]);
  float ex = expf(e - keyf32(emaxk[d * H + h]));
  float alpha = ex / fmaxf(den[d * H + h], 1e-16f);
  attn[gid] = alpha;                      // [T, H] row-major == gid
  const float w = alpha * (1.f / (float)H);
  const float* zs = z + (size_t)s * H * C + (size_t)h * C;
  float* hd = hm + (size_t)d * C;
  for (int c = 0; c < C; ++c) atomicAdd(&hd[c], w * zs[c]);
}

// Column sums / sums of squares for batch-norm.
__global__ void bn_stats(const float* __restrict__ x, int n, int C,
                         float* __restrict__ sum, float* __restrict__ sq) {
  int c = threadIdx.x;
  if (c >= C) return;
  float s = 0.f, s2 = 0.f;
  for (int r = blockIdx.x; r < n; r += gridDim.x) {
    float v = x[(size_t)r * C + c];
    s += v; s2 += v * v;
  }
  atomicAdd(&sum[c], s);
  atomicAdd(&sq[c], s2);
}

// out = relu( (x-mu)*rsqrt(var+eps)*g + b  [+ res] )
__global__ void bn_apply(const float* __restrict__ x,
                         const float* __restrict__ sum, const float* __restrict__ sq,
                         const float* __restrict__ g, const float* __restrict__ b,
                         const float* __restrict__ res, float* __restrict__ outp,
                         int n, int C) {
  int i = blockIdx.x * blockDim.x + threadIdx.x;
  if (i >= n * C) return;
  int c = i % C;
  float inv_n = 1.f / (float)n;
  float mu  = sum[c] * inv_n;
  float var = sq[c] * inv_n - mu * mu;
  float v = (x[i] - mu) * rsqrtf(var + 1e-5f) * g[c] + b[c];
  if (res) v += res[i];
  outp[i] = fmaxf(v, 0.f);
}

// TopK pooling: one 512-thread block per graph; bitonic sort of
// (tanh(x.p/||p||), idx) descending (ties -> lower index, matching lax.top_k),
// then gate & compact the kept nodes.
__global__ void topk_pool(const float* __restrict__ h, const float* __restrict__ p,
                          int dim, int npg, int k,
                          float* __restrict__ xnew, int* __restrict__ keep,
                          int* __restrict__ newid) {
  __shared__ float ss[512];
  __shared__ int   si[512];
  const int g = blockIdx.x, t = threadIdx.x;
  // ||p|| (block reduction)
  ss[t] = (t < dim) ? p[t] * p[t] : 0.f;
  __syncthreads();
  for (int off = 256; off > 0; off >>= 1) {
    if (t < off) ss[t] += ss[t + off];
    __syncthreads();
  }
  const float pn = sqrtf(ss[0]);
  __syncthreads();
  // scores
  float sc = -INFINITY;
  if (t < npg) {
    const float* row = h + (size_t)(g * npg + t) * dim;
    float dot = 0.f;
    for (int c = 0; c < dim; ++c) dot += row[c] * p[c];
    sc = tanhf(dot / pn);
  }
  ss[t] = sc; si[t] = t;
  __syncthreads();
  // bitonic sort, descending
  for (int ks = 2; ks <= 512; ks <<= 1) {
    for (int j = ks >> 1; j > 0; j >>= 1) {
      int ixj = t ^ j;
      if (ixj > t) {
        float sa = ss[t], sb = ss[ixj];
        int   ia = si[t], ib = si[ixj];
        bool aFirst = (sa > sb) || (sa == sb && ia < ib);
        bool desc   = ((t & ks) == 0);
        if (desc ? !aFirst : aFirst) {
          ss[t] = sb; ss[ixj] = sa; si[t] = ib; si[ixj] = ia;
        }
      }
      __syncthreads();
    }
  }
  if (t < k) {
    const int node = si[t];
    const float s = ss[t];
    const int gsrc = g * npg + node;
    const int gdst = g * k + t;
    for (int c = 0; c < dim; ++c)
      xnew[(size_t)gdst * dim + c] = h[(size_t)gsrc * dim + c] * s;
    keep[gsrc]  = 1;
    newid[gsrc] = gdst;
  }
}

__global__ void edge_remap(const int* __restrict__ src, const int* __restrict__ dst,
                           const int* __restrict__ em,
                           const int* __restrict__ keep, const int* __restrict__ newid,
                           int* __restrict__ srcn, int* __restrict__ dstn,
                           int* __restrict__ emn, int E) {
  int i = blockIdx.x * blockDim.x + threadIdx.x;
  if (i >= E) return;
  int s = src[i], d = dst[i];
  int m = (em ? em[i] : 1) && keep[s] && keep[d];
  srcn[i] = m ? newid[s] : 0;
  dstn[i] = m ? newid[d] : 0;
  emn[i]  = m;
}

// ------------------------------ GCN -----------------------------------------
__global__ void gcn_deg(const int* __restrict__ dst, const int* __restrict__ em,
                        float* __restrict__ deg, int E) {
  int i = blockIdx.x * blockDim.x + threadIdx.x;
  if (i >= E) return;
  if (em ? em[i] : 1) atomicAdd(&deg[dst[i]], 1.f);
}
__global__ void rsqrt_k(const float* __restrict__ deg, float* __restrict__ dinv, int n) {
  int i = blockIdx.x * blockDim.x + threadIdx.x;
  if (i < n) dinv[i] = (deg[i] > 0.f) ? rsqrtf(deg[i]) : 0.f;
}
// self-loop contribution: agg = xw * dinv^2  (gcn bias cancelled by group BN)
__global__ void gcn_self(const float* __restrict__ xw, const float* __restrict__ dinv,
                         float* __restrict__ agg, int n, int C) {
  int i = blockIdx.x * blockDim.x + threadIdx.x;
  if (i >= n * C) return;
  float di = dinv[i / C];
  agg[i] = xw[i] * di * di;
}
__global__ void gcn_edge(const int* __restrict__ src, const int* __restrict__ dst,
                         const int* __restrict__ em, const float* __restrict__ dinv,
                         const float* __restrict__ xw, float* __restrict__ agg,
                         int E, int C) {
  int i = blockIdx.x * blockDim.x + threadIdx.x;
  if (i >= E) return;
  if (!(em ? em[i] : 1)) return;
  int s = src[i], d = dst[i];
  float w = dinv[s] * dinv[d];
  const float* xs = xw + (size_t)s * C;
  float* ad = agg + (size_t)d * C;
  for (int c = 0; c < C; ++c) atomicAdd(&ad[c], w * xs[c]);
}

// Per-graph mean pooling: nodes of graph g are contiguous [g*k, (g+1)*k).
__global__ void mean_pool(const float* __restrict__ h, float* __restrict__ out,
                          int k, int C) {
  int g = blockIdx.x, c = threadIdx.x;
  if (c >= C) return;
  float s = 0.f;
  for (int i = 0; i < k; ++i) s += h[(size_t)(g * k + i) * C + c];
  out[g * C + c] = s / (float)k;
}

// ---------------------------------------------------------------------------
extern "C" void kernel_launch(void* const* d_in, const int* in_sizes, int n_in,
                              void* d_out, int out_size, void* d_ws, size_t ws_size,
                              hipStream_t stream) {
  (void)in_sizes; (void)n_in; (void)out_size; (void)ws_size;

  // Inputs in setup_inputs() insertion order: x, edge_index, batch, then params
  // in _init_params insertion order.
  const float* x  = (const float*)d_in[0];
  const int*   ei = (const int*)d_in[1];     // [2, E]: src row then dst row
  auto P = [&](int i) { return (const float*)d_in[i]; };
  const float* gatW[3] = {P(3),  P(11), P(19)};
  const float* gas [3] = {P(4),  P(12), P(20)};
  const float* gad [3] = {P(5),  P(13), P(21)};
  const float* bng [3] = {P(7),  P(15), P(23)};
  const float* bnb [3] = {P(8),  P(16), P(24)};
  const float* skW [3] = {P(9),  P(17), P(25)};
  const float* skb [3] = {P(10), P(18), P(26)};
  const float* poolp[2] = {P(27), P(28)};
  const float* gcnW = P(29);
  const float* gng  = P(31);
  const float* gnb  = P(32);

  float* out = (float*)d_out;

  // ---- workspace carve (bump allocator, 256B aligned) ----
  char* wsb = (char*)d_ws;
  size_t off = 0;
  auto alloc = [&](size_t bytes) -> void* {
    off = (off + 255) & ~(size_t)255;
    void* p = wsb + off;
    off += bytes;
    return p;
  };
  float*    z     = (float*)alloc((size_t)NNODES * 512 * 4);
  float*    hm    = (float*)alloc((size_t)NNODES * 128 * 4);
  float*    res   = (float*)alloc((size_t)NNODES * 128 * 4);
  float*    hA    = (float*)alloc((size_t)NNODES * 128 * 4);  // layer output
  float*    hB    = (float*)alloc((size_t)NNODES * 128 * 4);  // pooled features
  _Float16* Ah    = (_Float16*)alloc((size_t)NNODES * KPAD0 * 2);  // f16 GEMM A
  _Float16* Wt    = (_Float16*)alloc((size_t)512 * KPAD0 * 2);     // f16 GEMM B^T
  float*    asrc  = (float*)alloc((size_t)NNODES * HEADS * 4);
  float*    adst  = (float*)alloc((size_t)NNODES * HEADS * 4);
  unsigned* emaxk = (unsigned*)alloc((size_t)NNODES * HEADS * 4);
  float*    den   = (float*)alloc((size_t)NNODES * HEADS * 4);
  float*    bsum  = (float*)alloc(128 * 4);
  float*    bsq   = (float*)alloc(128 * 4);
  int*      keep  = (int*)alloc((size_t)NNODES * 4);
  int*      newid = (int*)alloc((size_t)NNODES * 4);
  int*      srcA  = (int*)alloc((size_t)NEDGE * 4);
  int*      dstA  = (int*)alloc((size_t)NEDGE * 4);
  int*      srcB  = (int*)alloc((size_t)NEDGE * 4);
  int*      dstB  = (int*)alloc((size_t)NEDGE * 4);
  int*      em1   = (int*)alloc((size_t)NEDGE * 4);
  int*      em2   = (int*)alloc((size_t)NEDGE * 4);
  float*    deg   = (float*)alloc((size_t)NNODES * 4);
  float*    dinvv = (float*)alloc((size_t)NNODES * 4);
  float*    xw    = (float*)alloc((size_t)BGRAPH * KP2 * 32 * 4);
  float*    agg   = (float*)alloc((size_t)BGRAPH * KP2 * 32 * 4);

  const int nArr[3]    = {NNODES, BGRAPH * KP1, BGRAPH * KP2};
  const int cinArr[3]  = {DINF, 128, 64};
  const int kpArr[3]   = {KPAD0, 128, 64};     // padded K per layer
  const int coutArr[3] = {128, 64, 32};
  const int npgArr[2]  = {NPG0, KP1};
  const int kArr[2]    = {KP1, KP2};

  // d_out layout (pytree of (pooled, {gnn_layer_0..2})):
  size_t attnOff[3];
  attnOff[0] = (size_t)BGRAPH * 32;
  attnOff[1] = attnOff[0] + (size_t)(NEDGE + nArr[0]) * HEADS;
  attnOff[2] = attnOff[1] + (size_t)(NEDGE + nArr[1]) * HEADS;

  const float* hin   = x;
  const int* curSrc  = ei;
  const int* curDst  = ei + NEDGE;
  const int* curMask = nullptr;                 // level 0: all edges valid

  for (int l = 0; l < 3; ++l) {
    const int n = nArr[l], Ci = cinArr[l], Kp = kpArr[l];
    const int Co = coutArr[l], HC = HEADS * Co;

    // Convert activations once per layer; weights per GEMM (stream-ordered).
    cvt_a_f16<<<CDIV(n * Kp, 256), 256, 0, stream>>>(hin, Ah, n, Ci, Kp);

    // z = hin @ gatW   (WMMA)
    cvt_w_f16_t<<<CDIV(HC * Kp, 256), 256, 0, stream>>>(gatW[l], Wt, Ci, HC, Kp);
    wmma_gemm_f16<<<dim3(HC / 32, n / 16), 32, 0, stream>>>(Ah, Wt, nullptr, z, n, HC, Kp);
    // res = hin @ skipW + skip_b   (WMMA)
    cvt_w_f16_t<<<CDIV(Co * Kp, 256), 256, 0, stream>>>(skW[l], Wt, Ci, Co, Kp);
    wmma_gemm_f16<<<dim3(Co / 32, n / 16), 32, 0, stream>>>(Ah, Wt, skb[l], res, n, Co, Kp);

    attn_coef<<<CDIV(n * HEADS, 256), 256, 0, stream>>>(z, gas[l], gad[l], asrc, adst, n, HEADS, Co);

    fill_u32<<<CDIV(n * HEADS, 256), 256, 0, stream>>>(emaxk, 0u, n * HEADS);
    fill_f32<<<CDIV(n * HEADS, 256), 256, 0, stream>>>(den, 0.f, n * HEADS);
    fill_f32<<<CDIV(n * Co, 256), 256, 0, stream>>>(hm, 0.f, n * Co);

    const int T = NEDGE + n;
    gat_edge_max   <<<CDIV(T * HEADS, 256), 256, 0, stream>>>(curSrc, curDst, curMask, asrc, adst, emaxk, NEDGE, n, HEADS);
    gat_edge_expsum<<<CDIV(T * HEADS, 256), 256, 0, stream>>>(curSrc, curDst, curMask, asrc, adst, emaxk, den, NEDGE, n, HEADS);
    gat_edge_aggr  <<<CDIV(T * HEADS, 256), 256, 0, stream>>>(curSrc, curDst, curMask, asrc, adst, emaxk, den, z, hm,
                                                              out + attnOff[l], NEDGE, n, HEADS, Co);

    fill_f32<<<1, 256, 0, stream>>>(bsum, 0.f, Co);
    fill_f32<<<1, 256, 0, stream>>>(bsq, 0.f, Co);
    bn_stats<<<256, Co, 0, stream>>>(hm, n, Co, bsum, bsq);
    bn_apply<<<CDIV(n * Co, 256), 256, 0, stream>>>(hm, bsum, bsq, bng[l], bnb[l], res, hA, n, Co);

    if (l < 2) {
      fill_i32<<<CDIV(n, 256), 256, 0, stream>>>(keep, 0, n);
      fill_i32<<<CDIV(n, 256), 256, 0, stream>>>(newid, 0, n);
      topk_pool<<<BGRAPH, 512, 0, stream>>>(hA, poolp[l], Co, npgArr[l], kArr[l], hB, keep, newid);
      int* nSrc  = (l == 0) ? srcA : srcB;
      int* nDst  = (l == 0) ? dstA : dstB;
      int* nMask = (l == 0) ? em1 : em2;
      edge_remap<<<CDIV(NEDGE, 256), 256, 0, stream>>>(curSrc, curDst, curMask, keep, newid, nSrc, nDst, nMask, NEDGE);
      curSrc = nSrc; curDst = nDst; curMask = nMask;
      hin = hB;
    }
  }

  // ---- GCN on the twice-pooled graph (n2 = 7872, C = 32) ----
  const int n2 = BGRAPH * KP2, Cg = 32;
  cvt_a_f16<<<CDIV(n2 * Cg, 256), 256, 0, stream>>>(hA, Ah, n2, Cg, Cg);
  cvt_w_f16_t<<<CDIV(Cg * Cg, 256), 256, 0, stream>>>(gcnW, Wt, Cg, Cg, Cg);
  wmma_gemm_f16<<<dim3(Cg / 32, n2 / 16), 32, 0, stream>>>(Ah, Wt, nullptr, xw, n2, Cg, Cg);

  fill_f32<<<CDIV(n2, 256), 256, 0, stream>>>(deg, 1.f, n2);   // self loop
  gcn_deg<<<CDIV(NEDGE, 256), 256, 0, stream>>>(curDst, curMask, deg, NEDGE);
  rsqrt_k<<<CDIV(n2, 256), 256, 0, stream>>>(deg, dinvv, n2);
  gcn_self<<<CDIV(n2 * Cg, 256), 256, 0, stream>>>(xw, dinvv, agg, n2, Cg);
  gcn_edge<<<CDIV(NEDGE, 256), 256, 0, stream>>>(curSrc, curDst, curMask, dinvv, xw, agg, NEDGE, Cg);

  fill_f32<<<1, 256, 0, stream>>>(bsum, 0.f, Cg);
  fill_f32<<<1, 256, 0, stream>>>(bsq, 0.f, Cg);
  bn_stats<<<256, Cg, 0, stream>>>(agg, n2, Cg, bsum, bsq);
  bn_apply<<<CDIV(n2 * Cg, 256), 256, 0, stream>>>(agg, bsum, bsq, gng, gnb, nullptr, hB, n2, Cg);

  mean_pool<<<BGRAPH, 32, 0, stream>>>(hB, out, KP2, Cg);
}
